// MultiHeadAttention_15599321219695
// MI455X (gfx1250) — matrix operand order repaired and checked by reading, not compiled, not run on hardware
//
#include <hip/hip_runtime.h>

#define D_MODEL 1024
#define N_HEAD  16
#define D_K     64
#define BATCH   2
#define SEQ     2048
#define NROWS   (BATCH * SEQ)

typedef __attribute__((ext_vector_type(16))) __bf16 v16bf;
typedef __attribute__((ext_vector_type(8)))  float  v8f;
typedef __attribute__((ext_vector_type(4)))  unsigned int u32x4;
typedef __attribute__((ext_vector_type(8)))  int i32x8;
typedef __attribute__((ext_vector_type(4)))  int i32x4;

union FragA {
    v16bf v;
    uint4 q[2];
};

static __device__ __forceinline__ unsigned short f32_to_bf16(float f) {
    unsigned u = __float_as_uint(f);
    unsigned r = u + 0x7FFFu + ((u >> 16) & 1u);  // round-to-nearest-even
    return (unsigned short)(r >> 16);
}

// ---- DPP row_ror butterfly reduce over a 16-lane row (all VALU, no LDS) ----
#define DPP_ROR_F(x, n) \
    __int_as_float(__builtin_amdgcn_update_dpp(0, __float_as_int(x), 0x120 + (n), 0xF, 0xF, true))

static __device__ __forceinline__ float rowmax16(float x) {
    x = fmaxf(x, DPP_ROR_F(x, 8));
    x = fmaxf(x, DPP_ROR_F(x, 4));
    x = fmaxf(x, DPP_ROR_F(x, 2));
    x = fmaxf(x, DPP_ROR_F(x, 1));
    return x;
}
static __device__ __forceinline__ float rowsum16(float x) {
    x += DPP_ROR_F(x, 8);
    x += DPP_ROR_F(x, 4);
    x += DPP_ROR_F(x, 2);
    x += DPP_ROR_F(x, 1);
    return x;
}

// ---- CDNA5 async global->LDS copy (ASYNCcnt-tracked, no VGPR data) ----
static __device__ __forceinline__ void async_load_b128(unsigned lds_byte_off,
                                                       const void* gaddr) {
    asm volatile("global_load_async_to_lds_b128 %0, %1, off"
                 :: "v"(lds_byte_off),
                    "v"((unsigned long long)(uintptr_t)gaddr)
                 : "memory");
}
static __device__ __forceinline__ void wait_async0() {
    asm volatile("s_wait_asynccnt 0x0" ::: "memory");
}
static __device__ __forceinline__ unsigned lds_off(const void* p) {
    return (unsigned)(uintptr_t)p;  // low 32 bits of generic shared addr = LDS offset
}

// ---- CDNA5 Tensor Data Mover: 2D tile Global->LDS (TENSORcnt-tracked) ----
// Builds a D# per CDNA5 ISA: count=1, data_size=2B,
// tile rows x rowlen elems, row stride in elems; groups 2/3 zero (<=2D).
static __device__ __forceinline__ void tdm_load_2d(
    unsigned ldsOff, const void* gptr, unsigned tile_rows, unsigned tile_rowlen,
    unsigned tensor_rows, unsigned row_stride_elems) {
    unsigned long long ga = (unsigned long long)(uintptr_t)gptr;
    u32x4 g0;
    g0[0] = 1u;                                   // count=1, user descriptor
    g0[1] = ldsOff;                               // lds_addr (bytes)
    g0[2] = (unsigned)ga;                         // global_addr[31:0]
    g0[3] = (unsigned)((ga >> 32) & 0x01FFFFFFu)  // global_addr[56:32]
            | 0x80000000u;                        // type = 2 ("image")
    i32x8 g1;
    g1[0] = 0x00010000;                                   // data_size = 1 (2 bytes)
    g1[1] = (int)((row_stride_elems & 0xFFFFu) << 16);    // tensor_dim0[15:0]
    g1[2] = (int)(((row_stride_elems >> 16) & 0xFFFFu) |
                  ((tensor_rows & 0xFFFFu) << 16));       // dim0 hi | tensor_dim1 lo
    g1[3] = (int)(((tensor_rows >> 16) & 0xFFFFu) |
                  (tile_rowlen << 16));                   // dim1 hi | tile_dim0
    g1[4] = (int)(tile_rows & 0xFFFFu);                   // tile_dim1 (tile_dim2=0)
    g1[5] = (int)row_stride_elems;                        // tensor_dim0_stride[31:0]
    g1[6] = 0;
    g1[7] = 0;
    i32x4 g2 = {0, 0, 0, 0};
    i32x4 g3 = {0, 0, 0, 0};
#if defined(__clang_major__) && (__clang_major__ >= 23)
    i32x8 g4 = {0, 0, 0, 0, 0, 0, 0, 0};
    __builtin_amdgcn_tensor_load_to_lds(g0, g1, g2, g3, g4, 0);
#else
    __builtin_amdgcn_tensor_load_to_lds(g0, g1, g2, g3, 0);
#endif
}
static __device__ __forceinline__ void wait_tensor0() {
    __builtin_amdgcn_s_wait_tensorcnt(0);
}

// ---------------------------------------------------------------------------
// fp32 -> bf16 conversion (8 elems / thread, vectorized)
// ---------------------------------------------------------------------------
__global__ __launch_bounds__(256) void cvt_f32_bf16_kernel(
    const float* __restrict__ in, unsigned short* __restrict__ out, int n) {
    int i = (blockIdx.x * 256 + threadIdx.x) * 8;
    if (i >= n) return;
    float4 a = *(const float4*)(in + i);
    float4 b = *(const float4*)(in + i + 4);
    union { unsigned short s[8]; uint4 q; } u;
    u.s[0] = f32_to_bf16(a.x); u.s[1] = f32_to_bf16(a.y);
    u.s[2] = f32_to_bf16(a.z); u.s[3] = f32_to_bf16(a.w);
    u.s[4] = f32_to_bf16(b.x); u.s[5] = f32_to_bf16(b.y);
    u.s[6] = f32_to_bf16(b.z); u.s[7] = f32_to_bf16(b.w);
    *(uint4*)(out + i) = u.q;
}

// ---------------------------------------------------------------------------
// Y[M,1024] = X[M,1024] @ W[1024,1024]^T + bias   (bf16 in, fp32 accum)
// 256 threads, 128x128 tile; K staged 64-wide through double-buffered LDS
// filled by the Tensor Data Mover (one tensor_load_to_lds per tile, issued
// by wave 0 (X) and wave 1 (W); fetch of k+1 runs under WMMA of k).
// Wave (of 8) computes 32x64 = 2x4 WMMA 16x16 tiles, 16 WMMA per K-step.
// ---------------------------------------------------------------------------
template <bool OUT_BF16>
__global__ __launch_bounds__(256) void gemm_bf16_kernel(
    const unsigned short* __restrict__ A,
    const unsigned short* __restrict__ W,
    const float* __restrict__ bias,
    void* __restrict__ outp, int M) {
    __shared__ unsigned short Xs[2][128 * 64];   // 16 KB per buffer
    __shared__ unsigned short Ws[2][128 * 64];

    const int tid   = threadIdx.x;
    const int lane  = tid & 31;
    const int wave  = tid >> 5;
    const int lh    = lane & 15;
    const int hi    = lane >> 4;
    const int waveM = wave >> 1;   // 0..3
    const int waveN = wave & 1;    // 0..1
    const int m0 = blockIdx.x * 128;
    const int n0 = blockIdx.y * 128;

    v8f acc[2][4];
#pragma unroll
    for (int i = 0; i < 2; ++i)
#pragma unroll
        for (int j = 0; j < 4; ++j)
            acc[i][j] = {};

    // TDM stage of 128x64 K-slice into buffer `buf`
    auto stage = [&](int buf, int k0) {
        if (wave == 0)
            tdm_load_2d(lds_off(&Xs[buf][0]), A + (size_t)m0 * D_MODEL + k0,
                        128, 64, (unsigned)M, D_MODEL);
        else if (wave == 1)
            tdm_load_2d(lds_off(&Ws[buf][0]), W + (size_t)n0 * D_MODEL + k0,
                        128, 64, D_MODEL, D_MODEL);
    };

    stage(0, 0);
    int cur = 0;
    for (int k0 = 0; k0 < D_MODEL; k0 += 64) {
        if (wave < 2) wait_tensor0();  // issuing waves: tile landed in LDS
        __syncthreads();               // all waves: data visible, prev buffer free
        if (k0 + 64 < D_MODEL) stage(cur ^ 1, k0 + 64);

#pragma unroll
        for (int kc = 0; kc < 2; ++kc) {   // two K=32 sub-steps of the 64-slice
            FragA a[2], b[4];
#pragma unroll
            for (int i = 0; i < 2; ++i) {
                int row  = waveM * 32 + i * 16 + lh;
                int base = row * 64 + kc * 32 + hi * 8;            // K 0..7 / 8..15
                a[i].q[0] = *(const uint4*)(&Xs[cur][base]);
                a[i].q[1] = *(const uint4*)(&Xs[cur][base + 16]);  // K 16..23 / 24..31
            }
#pragma unroll
            for (int j = 0; j < 4; ++j) {
                int row  = waveN * 64 + j * 16 + lh;
                int base = row * 64 + kc * 32 + hi * 8;            // K 0..7 / 8..15
                b[j].q[0] = *(const uint4*)(&Ws[cur][base]);
                b[j].q[1] = *(const uint4*)(&Ws[cur][base + 16]);  // K 16..23 / 24..31
            }
#pragma unroll
            for (int i = 0; i < 2; ++i)
#pragma unroll
                for (int j = 0; j < 4; ++j)
                    acc[i][j] = __builtin_amdgcn_wmma_f32_16x16x32_bf16(
                        false, a[i].v, false, b[j].v, (short)0, acc[i][j], false, false);
        }
        cur ^= 1;
    }

    // epilogue: bias + store
#pragma unroll
    for (int j = 0; j < 4; ++j) {
        int col = n0 + waveN * 64 + j * 16 + lh;
        float bj = bias[col];
#pragma unroll
        for (int i = 0; i < 2; ++i) {
#pragma unroll
            for (int r = 0; r < 8; ++r) {
                int row = m0 + waveM * 32 + i * 16 + r + hi * 8;
                float v = acc[i][j][r] + bj;
                if (OUT_BF16)
                    ((unsigned short*)outp)[(size_t)row * D_MODEL + col] = f32_to_bf16(v);
                else
                    ((float*)outp)[(size_t)row * D_MODEL + col] = v;
            }
        }
    }
}

// ---------------------------------------------------------------------------
// Flash attention: one (b,h, 64 q-rows) tile per 128-thread block.
// Each wave owns 16 q rows, full d_k=64 fp32 accumulator.
// Loop over 32 k-rows/iter: K tile async-staged, transposed V tile manual.
// ---------------------------------------------------------------------------
__global__ __launch_bounds__(128) void flash_attn_kernel(
    const unsigned short* __restrict__ Qp,
    const unsigned short* __restrict__ Kp,
    const unsigned short* __restrict__ Vp,
    const int* __restrict__ mask,
    unsigned short* __restrict__ attnout) {
    __shared__ unsigned short Kt[32 * 64];    // [krow][d]       4 KB
    __shared__ unsigned short Vt[64 * 40];    // [d][krow], pad  5 KB
    __shared__ unsigned short Pl[4][16 * 32]; // per-wave P tile 4 KB

    const int tid  = threadIdx.x;
    const int lane = tid & 31;
    const int wave = tid >> 5;
    const int lh   = lane & 15;
    const int hi   = lane >> 4;
    const int b    = blockIdx.y >> 4;
    const int h    = blockIdx.y & 15;
    const int q0   = blockIdx.x * 64 + wave * 16;

    // Q A-fragments (two K=32 chunks over d_k=64), loaded once
    FragA aQ[2];
    {
        size_t rowb = (size_t)(b * SEQ + q0 + lh) * D_MODEL;
#pragma unroll
        for (int c = 0; c < 2; ++c) {
            int col = h * D_K + c * 32 + hi * 8;
            aQ[c].q[0] = *(const uint4*)(Qp + rowb + col);
            aQ[c].q[1] = *(const uint4*)(Qp + rowb + col + 16);
        }
    }

    v8f o[4];
#pragma unroll
    for (int t = 0; t < 4; ++t) o[t] = {};
    float mstat[8], lstat[8];
#pragma unroll
    for (int j = 0; j < 8; ++j) { mstat[j] = -1e30f; lstat[j] = 0.f; }

    const int ldRow = tid >> 2;        // 0..31 (k-row)
    const int ldQd  = (tid & 3) * 16;  // d chunk

    for (int kt = 0; kt < SEQ / 32; ++kt) {
        // stage K (async, row-major) and V (manual transpose)
        {
            size_t grow = (size_t)(b * SEQ + kt * 32 + ldRow) * D_MODEL + h * D_K + ldQd;
            unsigned lk = lds_off(&Kt[ldRow * 64 + ldQd]);
            async_load_b128(lk,      Kp + grow);
            async_load_b128(lk + 16, Kp + grow + 8);
            union { uint4 q[2]; unsigned short s[16]; } vv;
            vv.q[0] = *(const uint4*)(Vp + grow);
            vv.q[1] = *(const uint4*)(Vp + grow + 8);
#pragma unroll
            for (int e = 0; e < 16; ++e)
                Vt[(ldQd + e) * 40 + ldRow] = vv.s[e];
        }
        wait_async0();
        __syncthreads();

        // scores: 16 q-rows x 32 k-rows, K-dim = 64 (2 chunks)
        v8f s0 = {}, s1 = {};
#pragma unroll
        for (int c = 0; c < 2; ++c) {
            FragA bK0, bK1;
            int base0 = (0 * 16 + lh) * 64 + c * 32 + hi * 8;
            int base1 = (1 * 16 + lh) * 64 + c * 32 + hi * 8;
            bK0.q[0] = *(const uint4*)(&Kt[base0]);
            bK0.q[1] = *(const uint4*)(&Kt[base0 + 16]);
            bK1.q[0] = *(const uint4*)(&Kt[base1]);
            bK1.q[1] = *(const uint4*)(&Kt[base1 + 16]);
            s0 = __builtin_amdgcn_wmma_f32_16x16x32_bf16(false, aQ[c].v, false, bK0.v,
                                                         (short)0, s0, false, false);
            s1 = __builtin_amdgcn_wmma_f32_16x16x32_bf16(false, aQ[c].v, false, bK1.v,
                                                         (short)0, s1, false, false);
        }

        // scale, mask, online softmax (DPP reductions across the 16-lane row)
        float p0[8], p1[8];
        const size_t mbase = (size_t)b * SEQ * SEQ + (size_t)kt * 32;
#pragma unroll
        for (int j = 0; j < 8; ++j) {
            int qrow = q0 + j + hi * 8;
            const int* mp = mask + mbase + (size_t)qrow * SEQ;
            float v0 = s0[j] * 0.125f;   // 1/sqrt(64)
            float v1 = s1[j] * 0.125f;
            if (mp[lh] == 0)      v0 = -1e9f;
            if (mp[16 + lh] == 0) v1 = -1e9f;
            float mx    = rowmax16(fmaxf(v0, v1));
            float mnew  = fmaxf(mstat[j], mx);
            float alpha = __expf(mstat[j] - mnew);
            mstat[j] = mnew;
            p0[j] = __expf(v0 - mnew);
            p1[j] = __expf(v1 - mnew);
            float rs = rowsum16(p0[j] + p1[j]);
            lstat[j] = lstat[j] * alpha + rs;
#pragma unroll
            for (int t = 0; t < 4; ++t) o[t][j] *= alpha;
        }

        // P (C layout) -> LDS -> A layout, then PV WMMA
        unsigned short* pw = &Pl[wave][0];
#pragma unroll
        for (int j = 0; j < 8; ++j) {
            pw[(j + hi * 8) * 32 + lh]      = f32_to_bf16(p0[j]);
            pw[(j + hi * 8) * 32 + 16 + lh] = f32_to_bf16(p1[j]);
        }
        asm volatile("s_wait_dscnt 0x0" ::: "memory");  // same-wave LDS RAW
        FragA aP;
        {
            int base = lh * 32 + hi * 8;
            aP.q[0] = *(const uint4*)(&pw[base]);
            aP.q[1] = *(const uint4*)(&pw[base + 16]);
        }
#pragma unroll
        for (int t = 0; t < 4; ++t) {
            FragA bV;
            int base = (t * 16 + lh) * 40 + hi * 8;
            bV.q[0] = *(const uint4*)(&Vt[base]);
            bV.q[1] = *(const uint4*)(&Vt[base + 16]);
            o[t] = __builtin_amdgcn_wmma_f32_16x16x32_bf16(false, aP.v, false, bV.v,
                                                           (short)0, o[t], false, false);
        }
        __syncthreads();
    }

    // normalize + store bf16 (heads concatenated)
#pragma unroll
    for (int j = 0; j < 8; ++j) {
        float inv = 1.0f / lstat[j];
        size_t rowb = (size_t)(b * SEQ + q0 + j + hi * 8) * D_MODEL;
#pragma unroll
        for (int t = 0; t < 4; ++t)
            attnout[rowb + h * D_K + t * 16 + lh] = f32_to_bf16(o[t][j] * inv);
    }
}

// ---------------------------------------------------------------------------
// Host orchestration
// ---------------------------------------------------------------------------
extern "C" void kernel_launch(void* const* d_in, const int* in_sizes, int n_in,
                              void* d_out, int out_size, void* d_ws, size_t ws_size,
                              hipStream_t stream) {
    const float* q   = (const float*)d_in[0];
    const float* k   = (const float*)d_in[1];
    const float* v   = (const float*)d_in[2];
    const int* mask  = (const int*)d_in[3];
    const float* Wq  = (const float*)d_in[4];
    const float* bq  = (const float*)d_in[5];
    const float* Wk  = (const float*)d_in[6];
    const float* bk  = (const float*)d_in[7];
    const float* Wv  = (const float*)d_in[8];
    const float* bv  = (const float*)d_in[9];
    const float* Wo  = (const float*)d_in[10];
    const float* bo  = (const float*)d_in[11];

    const size_t MB = 1024ull * 1024ull;
    char* ws = (char*)d_ws;
    unsigned short* qb   = (unsigned short*)(ws + 0 * MB);   // 8 MB each
    unsigned short* kb   = (unsigned short*)(ws + 8 * MB);
    unsigned short* vb   = (unsigned short*)(ws + 16 * MB);
    unsigned short* Wqb  = (unsigned short*)(ws + 24 * MB);  // 2 MB each
    unsigned short* Wkb  = (unsigned short*)(ws + 26 * MB);
    unsigned short* Wvb  = (unsigned short*)(ws + 28 * MB);
    unsigned short* Wob  = (unsigned short*)(ws + 30 * MB);
    unsigned short* Qp   = (unsigned short*)(ws + 32 * MB);  // 8 MB each
    unsigned short* Kp   = (unsigned short*)(ws + 40 * MB);
    unsigned short* Vp   = (unsigned short*)(ws + 48 * MB);
    unsigned short* attn = (unsigned short*)(ws + 56 * MB);

    const int nAct = NROWS * D_MODEL;   // 4M elems
    const int nWgt = D_MODEL * D_MODEL; // 1M elems

    cvt_f32_bf16_kernel<<<nAct / 2048, 256, 0, stream>>>(q, qb, nAct);
    cvt_f32_bf16_kernel<<<nAct / 2048, 256, 0, stream>>>(k, kb, nAct);
    cvt_f32_bf16_kernel<<<nAct / 2048, 256, 0, stream>>>(v, vb, nAct);
    cvt_f32_bf16_kernel<<<nWgt / 2048, 256, 0, stream>>>(Wq, Wqb, nWgt);
    cvt_f32_bf16_kernel<<<nWgt / 2048, 256, 0, stream>>>(Wk, Wkb, nWgt);
    cvt_f32_bf16_kernel<<<nWgt / 2048, 256, 0, stream>>>(Wv, Wvb, nWgt);
    cvt_f32_bf16_kernel<<<nWgt / 2048, 256, 0, stream>>>(Wo, Wob, nWgt);

    dim3 ggrid(NROWS / 128, D_MODEL / 128);
    gemm_bf16_kernel<true><<<ggrid, 256, 0, stream>>>(qb, Wqb, bq, (void*)Qp, NROWS);
    gemm_bf16_kernel<true><<<ggrid, 256, 0, stream>>>(kb, Wkb, bk, (void*)Kp, NROWS);
    gemm_bf16_kernel<true><<<ggrid, 256, 0, stream>>>(vb, Wvb, bv, (void*)Vp, NROWS);

    dim3 agrid(SEQ / 64, BATCH * N_HEAD);
    flash_attn_kernel<<<agrid, 128, 0, stream>>>(Qp, Kp, Vp, mask, attn);

    gemm_bf16_kernel<false><<<ggrid, 256, 0, stream>>>(attn, Wob, bo, d_out, NROWS);
}
